// FreqPruningLTM_57226144252003
// MI455X (gfx1250) — matrix operand order repaired and checked by reading, not compiled, not run on hardware
//
#include <hip/hip_runtime.h>
#include <hip/hip_bf16.h>

// ---------------------------------------------------------------------------
// FreqPruningLTM on MI455X (gfx1250, wave32):
//   scores = Q @ K^T via v_wmma_f32_16x16x32_bf16 (f32 accum),
//   2 query tiles (32 queries) per wave sharing one streamed B tile,
//   double-buffered B loads, fused streaming top-50 with register-threshold
//   fast path, chunked partial top-k -> merge+softmax+gather kernel.
// ---------------------------------------------------------------------------

typedef __attribute__((ext_vector_type(16))) __bf16 v16bf;
typedef __attribute__((ext_vector_type(8)))  __bf16 v8bf;
typedef __attribute__((ext_vector_type(8)))  float  v8f;

#define DK        128      // key/query depth
#define DV        128      // value depth
#define KSEL      50       // top-k
#define NCHUNK    8        // key-space split for parallelism (32768/8 = 4096)
#define QT        32       // queries per wave (two 16-row WMMA tiles)
#define NEG_INF   (-__builtin_inff())

// ---------------------------------------------------------------------------
// Kernel 0: f32 -> bf16 (round to nearest even)
// ---------------------------------------------------------------------------
__global__ void cvt_bf16_kernel(const float* __restrict__ src,
                                unsigned short* __restrict__ dst, int n) {
  int i = blockIdx.x * blockDim.x + threadIdx.x;
  if (i < n) {
    unsigned int u = __float_as_uint(src[i]);
    unsigned int r = u + 0x7FFFu + ((u >> 16) & 1u);
    dst[i] = (unsigned short)(r >> 16);
  }
}

// ---------------------------------------------------------------------------
// Streaming top-50 maintenance for one 16x16 score tile held in C registers.
// rowOff selects which 16-row slice of the per-block top-k buffers to update.
// All branches are wave-uniform (ballot), so WMMA EXEC stays all-ones.
// ---------------------------------------------------------------------------
__device__ __forceinline__ void topk_update(
    const v8f& c, int keyTile, int rowOff,
    int lane, int l15, int rbase,
    float* __restrict__ tile, float* __restrict__ topS, int* __restrict__ topI,
    float* __restrict__ thrSh, float thr[8], float& minval, int& minpos)
{
  int anyc = 0;
  #pragma unroll
  for (int v = 0; v < 8; ++v) anyc |= (c[v] > thr[v]) ? 1 : 0;
  if (__ballot(anyc)) {                    // uniform (wave-wide) branch
    #pragma unroll
    for (int v = 0; v < 8; ++v) tile[(rbase + v) * 17 + l15] = c[v];
    __syncthreads();
    if (lane < 16) {                       // lane owns score-tile row `lane`
      const int row = rowOff + lane;
      float* ts = &topS[row * KSEL];
      int*   ti = &topI[row * KSEL];
      for (int n = 0; n < 16; ++n) {
        float s = tile[lane * 17 + n];
        if (s > minval) {                  // strict: keeps earliest on ties
          ts[minpos] = s;
          ti[minpos] = keyTile + n;
          minval = ts[0]; minpos = 0;      // rescan for new min-of-50
          for (int j = 1; j < KSEL; ++j) {
            float x = ts[j];
            if (x < minval) { minval = x; minpos = j; }
          }
        }
      }
      thrSh[row] = minval;
    }
    __syncthreads();
    #pragma unroll
    for (int v = 0; v < 8; ++v) thr[v] = thrSh[rowOff + rbase + v];
  }
}

// ---------------------------------------------------------------------------
// Kernel 1: per-(32-query tile, key chunk) WMMA scores + streaming top-50.
// One wave32 per block. Grid: (numQueries/QT, NCHUNK).
// ---------------------------------------------------------------------------
__global__ __launch_bounds__(32)
void scores_topk_kernel(const __bf16* __restrict__ WQ,
                        const __bf16* __restrict__ WK,
                        float* __restrict__ candS,
                        int*   __restrict__ candI,
                        int chunkKeys) {
  __shared__ float tile[16 * 17];          // one 16x16 score tile, padded
  __shared__ float topS[QT * KSEL];        // per-row unsorted top-50 scores
  __shared__ int   topI[QT * KSEL];        // per-row key indices
  __shared__ float thrSh[QT];              // per-row current min-of-top50

  const int lane   = threadIdx.x;          // 0..31 (wave32)
  const int l15    = lane & 15;
  const int half   = lane >> 4;
  const int rbase  = half * 8;             // rows this lane's C VGPRs cover
  const int qBase  = blockIdx.x * QT;
  const int chunk  = blockIdx.y;
  const int keyBase = chunk * chunkKeys;

  if (lane < 16) {
    for (int r = lane; r < QT; r += 16) {
      for (int j = 0; j < KSEL; ++j) {
        topS[r * KSEL + j] = NEG_INF;
        topI[r * KSEL + j] = 0;
      }
      thrSh[r] = NEG_INF;
    }
  }
  __syncthreads();

  // --- Load A tiles (2 x 16x128 queries) once; ISA 16-bit A layout:
  //     lane half 0: K = {b..b+7, b+16..b+23} with b = 32*cc,
  //     lane half 1: same shifted by +8.
  v16bf a0[4], a1[4];
  {
    const __bf16* q0 = WQ + (size_t)(qBase + l15) * DK;
    const __bf16* q1 = q0 + 16 * DK;
    #pragma unroll
    for (int cc = 0; cc < 4; ++cc) {
      const int base = cc * 32 + half * 8;
      v8bf lo0 = *(const v8bf*)(q0 + base);
      v8bf hi0 = *(const v8bf*)(q0 + base + 16);
      v8bf lo1 = *(const v8bf*)(q1 + base);
      v8bf hi1 = *(const v8bf*)(q1 + base + 16);
      #pragma unroll
      for (int i = 0; i < 8; ++i) {
        a0[cc][i] = lo0[i]; a0[cc][i + 8] = hi0[i];
        a1[cc][i] = lo1[i]; a1[cc][i + 8] = hi1[i];
      }
    }
  }

  // Per-lane threshold registers for the rows each C accumulator covers.
  float thr0[8], thr1[8];
  #pragma unroll
  for (int v = 0; v < 8; ++v) { thr0[v] = NEG_INF; thr1[v] = NEG_INF; }
  float min0 = NEG_INF, min1 = NEG_INF;    // valid for lanes < 16 (row owners)
  int   pos0 = 0, pos1 = 0;

  const int numTiles = chunkKeys / 16;

  // B tile loader: lane holds key column n=l15; same K order as A
  // (half 0 -> K = 32*cc + {0..7, 16..23}, half 1 -> same shifted by +8).
  auto loadB = [&](int t, v16bf (&bd)[4]) {
    const __bf16* krow = WK + (size_t)(keyBase + t * 16 + l15) * DK + half * 8;
    #pragma unroll
    for (int cc = 0; cc < 4; ++cc) {
      v8bf lo = *(const v8bf*)(krow + cc * 32);
      v8bf hi = *(const v8bf*)(krow + cc * 32 + 16);
      #pragma unroll
      for (int i = 0; i < 8; ++i) { bd[cc][i] = lo[i]; bd[cc][i + 8] = hi[i]; }
    }
    if (t + 2 < numTiles) {                // prefetch 2 tiles ahead:
      const __bf16* prow =                 // 32 lanes cover 16 rows x 2 lines
          WK + (size_t)(keyBase + (t + 2) * 16 + l15) * DK + half * 64;
      __builtin_prefetch(prow, 0, 3);
    }
  };

  auto compute = [&](const v16bf (&b)[4], int keyTile) {
    v8f c0 = {}, c1 = {};
    #pragma unroll
    for (int cc = 0; cc < 4; ++cc) {       // two independent accumulator chains
      c0 = __builtin_amdgcn_wmma_f32_16x16x32_bf16(
               false, a0[cc], false, b[cc], (short)0, c0, false, false);
      c1 = __builtin_amdgcn_wmma_f32_16x16x32_bf16(
               false, a1[cc], false, b[cc], (short)0, c1, false, false);
    }
    topk_update(c0, keyTile, 0,  lane, l15, rbase, tile, topS, topI, thrSh,
                thr0, min0, pos0);
    topk_update(c1, keyTile, 16, lane, l15, rbase, tile, topS, topI, thrSh,
                thr1, min1, pos1);
  };

  // Software-pipelined main loop: B loads for tile t+1 are in flight while
  // tile t's WMMAs execute (explicit double buffer).
  v16bf b0[4], b1[4];
  loadB(0, b0);
  for (int t = 0; t < numTiles; t += 2) {
    if (t + 1 < numTiles) loadB(t + 1, b1);
    compute(b0, keyBase + t * 16);
    if (t + 1 < numTiles) {
      if (t + 2 < numTiles) loadB(t + 2, b0);
      compute(b1, keyBase + (t + 1) * 16);
    }
  }
  __syncthreads();

  // Emit this chunk's 50 candidates per row.
  if (lane < 16) {
    for (int r = lane; r < QT; r += 16) {
      const int q = qBase + r;
      float* cs = candS + ((size_t)q * NCHUNK + chunk) * KSEL;
      int*   ci = candI + ((size_t)q * NCHUNK + chunk) * KSEL;
      for (int j = 0; j < KSEL; ++j) {
        cs[j] = topS[r * KSEL + j];
        ci[j] = topI[r * KSEL + j];
      }
    }
  }
}

// ---------------------------------------------------------------------------
// Kernel 2: merge 8x50 candidates -> sorted top-50, softmax, gather V rows.
// One block (128 threads = 4 waves) per query.
// ---------------------------------------------------------------------------
#define NCAND (NCHUNK * KSEL)              // 400

__global__ __launch_bounds__(128)
void merge_kernel(const float* __restrict__ candS,
                  const int*   __restrict__ candI,
                  const float* __restrict__ values,
                  float* __restrict__ outW,
                  int*   __restrict__ outIdx,
                  float* __restrict__ outWt) {
  __shared__ float cs[NCAND];
  __shared__ int   ci[NCAND];
  __shared__ float redS[4];
  __shared__ int   redI[4];
  __shared__ int   redSlot[4];
  __shared__ float selS[KSEL];
  __shared__ int   selI[KSEL];
  __shared__ float w[KSEL];

  const int q   = blockIdx.x;
  const int tid = threadIdx.x;
  const int lane = tid & 31;
  const int wv   = tid >> 5;

  for (int i = tid; i < NCAND; i += 128) {
    cs[i] = candS[(size_t)q * NCAND + i];
    ci[i] = candI[(size_t)q * NCAND + i];
  }
  __syncthreads();

  // 50 rounds of parallel argmax -> descending order (jax top_k semantics,
  // ties broken toward the lower key index).
  for (int k = 0; k < KSEL; ++k) {
    float bs = NEG_INF; int bi = 0x7fffffff; int bslot = tid;
    for (int i = tid; i < NCAND; i += 128) {
      float s = cs[i]; int id = ci[i];
      if (s > bs || (s == bs && id < bi)) { bs = s; bi = id; bslot = i; }
    }
    #pragma unroll
    for (int off = 16; off > 0; off >>= 1) {
      float os = __shfl_down(bs, off, 32);
      int   oi = __shfl_down(bi, off, 32);
      int   ol = __shfl_down(bslot, off, 32);
      if (os > bs || (os == bs && oi < bi)) { bs = os; bi = oi; bslot = ol; }
    }
    if (lane == 0) { redS[wv] = bs; redI[wv] = bi; redSlot[wv] = bslot; }
    __syncthreads();
    if (tid == 0) {
      float s0 = redS[0]; int i0 = redI[0]; int sl = redSlot[0];
      #pragma unroll
      for (int j = 1; j < 4; ++j) {
        if (redS[j] > s0 || (redS[j] == s0 && redI[j] < i0)) {
          s0 = redS[j]; i0 = redI[j]; sl = redSlot[j];
        }
      }
      selS[k] = s0; selI[k] = i0;
      cs[sl] = NEG_INF;                    // remove winner
    }
    __syncthreads();
  }

  // Softmax over the 50 selected scores.
  if (tid == 0) {
    float m = selS[0];
    for (int j = 1; j < KSEL; ++j) m = fmaxf(m, selS[j]);
    float sum = 0.f;
    for (int j = 0; j < KSEL; ++j) { float e = expf(selS[j] - m); w[j] = e; sum += e; }
    float inv = 1.f / sum;
    for (int j = 0; j < KSEL; ++j) w[j] *= inv;
  }
  __syncthreads();

  if (tid < KSEL) {
    outIdx[(size_t)q * KSEL + tid] = selI[tid];
    outWt [(size_t)q * KSEL + tid] = w[tid];
  }

  // Weighted gather: thread = output channel d; V-row reads are coalesced.
  float acc = 0.f;
  for (int j = 0; j < KSEL; ++j)
    acc += w[j] * values[(size_t)selI[j] * DV + tid];
  outW[(size_t)q * DV + tid] = acc;
}

// ---------------------------------------------------------------------------
// Launcher
// ---------------------------------------------------------------------------
extern "C" void kernel_launch(void* const* d_in, const int* in_sizes, int n_in,
                              void* d_out, int out_size, void* d_ws, size_t ws_size,
                              hipStream_t stream) {
  const float* Q = (const float*)d_in[0];   // [B, 128]
  const float* K = (const float*)d_in[1];   // [M, 128]
  const float* V = (const float*)d_in[2];   // [M, 128]

  const int B = in_sizes[0] / DK;           // 8192
  const int M = in_sizes[1] / DK;           // 32768
  const int chunkKeys = M / NCHUNK;         // 4096

  // Workspace layout
  unsigned short* WQ = (unsigned short*)d_ws;                 // B*DK bf16
  unsigned short* WK = WQ + (size_t)B * DK;                   // M*DK bf16
  float* candS = (float*)(WK + (size_t)M * DK);               // B*NCAND f32
  int*   candI = (int*)(candS + (size_t)B * NCAND);           // B*NCAND i32

  // Output layout (tuple concatenated flat)
  float* outW   = (float*)d_out;                              // [B,128]
  int*   outIdx = (int*)(outW + (size_t)B * DV);              // [B,50]
  float* outWt  = (float*)(outIdx + (size_t)B * KSEL);        // [B,50]

  const int nq = B * DK, nk = M * DK;
  cvt_bf16_kernel<<<(nq + 255) / 256, 256, 0, stream>>>(Q, WQ, nq);
  cvt_bf16_kernel<<<(nk + 255) / 256, 256, 0, stream>>>(K, WK, nk);

  dim3 g1(B / QT, NCHUNK);
  scores_topk_kernel<<<g1, 32, 0, stream>>>((const __bf16*)WQ, (const __bf16*)WK,
                                            candS, candI, chunkKeys);

  merge_kernel<<<B, 128, 0, stream>>>(candS, candI, V, outW, outIdx, outWt);
}
